// SelfAtt_18743237280196
// MI455X (gfx1250) — matrix operand order repaired and checked by reading, not compiled, not run on hardware
//
#include <hip/hip_runtime.h>
#include <hip/hip_bf16.h>

typedef __attribute__((ext_vector_type(16))) __bf16        v16bf;
typedef __attribute__((ext_vector_type(8)))  float         v8f;
typedef __attribute__((ext_vector_type(8)))  unsigned int  v8u;
typedef __attribute__((ext_vector_type(4)))  unsigned int  v4u;

// Native f32 -> bf16 (v_cvt on gfx1250), bit pattern as ushort.
static __device__ __forceinline__ unsigned short f2bf(float f) {
  return __builtin_bit_cast(unsigned short, (__bf16)f);
}

static __device__ __forceinline__ v8u concat4(v4u lo, v4u hi) {
  union { v8u w; v4u q[2]; } u;
  u.q[0] = lo; u.q[1] = hi;
  return u.w;
}

static __device__ __forceinline__ v8f wmma_bf16(v8u a, v8u b, v8f c) {
  return __builtin_amdgcn_wmma_f32_16x16x32_bf16(
      false, __builtin_bit_cast(v16bf, a),
      false, __builtin_bit_cast(v16bf, b),
      (short)0, c, false, false);
}

// Async global->LDS 16B copy (ASYNCcnt path).
static __device__ __forceinline__ void async_copy_b128(unsigned ldsoff,
                                                       unsigned long long gaddr) {
  asm volatile("global_load_async_to_lds_b128 %0, %1, off"
               :: "v"(ldsoff), "v"(gaddr) : "memory");
}
static __device__ __forceinline__ void wait_async() {
  asm volatile("s_wait_asynccnt 0x0" ::: "memory");
}

// TDM: DMA a contiguous run of `elems` bf16 from global to LDS (TENSORcnt path).
// D# group0: count=1, lds_addr, global_addr[56:0], type=2.
// D# group1: data_size=2B, tensor_dim0=tile_dim0=elems, dim1=1, stride0=elems.
static __device__ __forceinline__ void tdm_load_row(unsigned ldsoff,
                                                    unsigned long long gaddr,
                                                    unsigned elems) {
  v4u g0;
  g0[0] = 1u;                                   // count=1 (valid user descriptor)
  g0[1] = ldsoff;                               // lds_addr (bytes)
  g0[2] = (unsigned)gaddr;                      // global_addr[31:0]
  g0[3] = ((unsigned)(gaddr >> 32) & 0x01FFFFFFu) | 0x80000000u;  // [56:32]|type=2
  v8u g1;
  g1[0] = 0x00010000u;                          // data_size=1 -> 2-byte elements
  g1[1] = (elems & 0xFFFFu) << 16;              // tensor_dim0[15:0]
  g1[2] = 0x00010000u | (elems >> 16);          // tensor_dim0[31:16], tensor_dim1=1
  g1[3] = (elems & 0xFFFFu) << 16;              // tile_dim0
  g1[4] = 0x00000001u;                          // tile_dim1=1
  g1[5] = elems;                                // tensor_dim0_stride[31:0]
  g1[6] = (elems & 0xFFFFu) << 16;              // tensor_dim1_stride[15:0]
  g1[7] = 0u;
  asm volatile("tensor_load_to_lds %0, %1" :: "s"(g0), "s"(g1) : "memory");
}

#define C_DIM 256
#define HW_DIM 4096
#define TILE_US 8192  // ushorts per 256x32 (or 32x256) bf16 tile = 16KB

// ---------------- Pass 0: pre-convert weights to bf16 ----------------
__global__ __launch_bounds__(256) void prep_w_kernel(
    const float* __restrict__ Wq, const float* __restrict__ Wk,
    const float* __restrict__ Wv, unsigned short* __restrict__ Wbf)
{
  int t = blockIdx.x * 256 + threadIdx.x;          // 0 .. 3*65536-1
  const float* src = (t < 65536) ? Wq : (t < 131072) ? Wk : Wv;
  Wbf[t] = f2bf(src[t & 65535]);
}

// Tile-image index helpers: each WMMA B fragment's 8 u32 words are contiguous.
// K image, element (c=0..255, j=0..31):
//   p=c>>1; uidx = (((p>>4)*2 + ((p>>3)&1))*32 + j)*8 + (p&7)
// V image, element (j=0..31, c=0..255):
//   p=j>>1; uidx = (c*2 + (p>>3))*8 + (p&7)
static __device__ __forceinline__ int kimg_u(int c, int j) {
  int p = c >> 1;
  return (((p >> 4) * 2 + ((p >> 3) & 1)) * 32 + j) * 8 + (p & 7);
}
static __device__ __forceinline__ int vimg_u(int j, int c) {
  int p = j >> 1;
  return (c * 2 + (p >> 3)) * 8 + (p & 7);
}

// ---------------- Pass 1: Q/K/V projection GEMMs (WMMA bf16) ----------------
__global__ __launch_bounds__(256) void proj_qkv_kernel(
    const float* __restrict__ x, const unsigned short* __restrict__ Wbf,
    const float* __restrict__ bq, const float* __restrict__ bk,
    const float* __restrict__ bv,
    unsigned short* __restrict__ Qt, unsigned short* __restrict__ Kbuf,
    unsigned short* __restrict__ Vbuf)
{
  const int n = blockIdx.y;
  const int lbase = blockIdx.x * 64;
  const int tid = threadIdx.x, lane = tid & 31, lrow = lane & 15,
            half = lane >> 4, wave = tid >> 5;

  // x tile (256 x 64) bf16, fragment-contiguous image:
  // element (c,l): p=c>>1; uidx = (((p>>4)*2 + ((p>>3)&1))*64 + l)*8 + (p&7)
  __shared__ alignas(16) unsigned short xs[256 * 64];
  const float* xn = x + (size_t)n * C_DIM * HW_DIM;
  for (int t = tid; t < 256 * 64; t += 256) {
    int c = t >> 6, l = t & 63;
    int p = c >> 1;
    int uidx = (((p >> 4) * 2 + ((p >> 3) & 1)) * 64 + l) * 8 + (p & 7);
    xs[uidx * 2 + (c & 1)] = f2bf(xn[(size_t)c * HW_DIM + lbase + l]);
  }
  __syncthreads();
  const v4u* xs4 = (const v4u*)xs;

  for (int item = wave; item < 192; item += 8) {
    const int mat = item >> 6, rem = item & 63, mt = rem >> 2, nt = rem & 3;
    const float* bm = (mat == 0) ? bq : (mat == 1) ? bk : bv;
    const int o0 = mt * 16, l0 = nt * 16;
    v8f acc = {0.f, 0.f, 0.f, 0.f, 0.f, 0.f, 0.f, 0.f};
    const v4u* wrow4 =
        (const v4u*)(Wbf + ((size_t)mat * 65536) + (size_t)(o0 + lrow) * C_DIM);
#pragma unroll
    for (int kt = 0; kt < 8; ++kt) {
      // A: two contiguous uint4 runs (v=0..3, v=4..7)
      v8u a = concat4(wrow4[kt * 4 + 2 * half], wrow4[kt * 4 + 2 * half + 1]);
      // B: fragment-contiguous image -> two ds_load_b128
      int fb = ((kt * 2 + half) * 64 + l0 + lrow) * 2;
      v8u b = concat4(xs4[fb], xs4[fb + 1]);
      acc = wmma_bf16(a, b, acc);
    }
#pragma unroll
    for (int v = 0; v < 8; ++v) {
      int o = o0 + v + 8 * half;            // C/D layout: M = v + 8*(lane/16)
      int l = lbase + l0 + lrow;            //             N = lane%16
      unsigned short hv = f2bf(acc[v] + bm[o]);
      if (mat == 0) {
        Qt[((size_t)n * HW_DIM + l) * C_DIM + o] = hv;
      } else {
        int tile = l >> 5, jj = l & 31;
        size_t base = ((size_t)n * 128 + tile) * TILE_US;
        if (mat == 1) Kbuf[base + kimg_u(o, jj) * 2 + (o & 1)] = hv;
        else          Vbuf[base + vimg_u(jj, o) * 2 + (jj & 1)] = hv;
      }
    }
  }
}

// ---------------- Pass 2: fused flash attention + residual ----------------
__global__ __launch_bounds__(256) void flash_att_kernel(
    const float* __restrict__ x, const float* __restrict__ gamma,
    const unsigned short* __restrict__ Qt, const unsigned short* __restrict__ Kbuf,
    const unsigned short* __restrict__ Vbuf, float* __restrict__ out)
{
  const int n = blockIdx.y;
  const int i0 = blockIdx.x * 128;
  const int tid = threadIdx.x, lane = tid & 31, lrow = lane & 15,
            half = lane >> 4, wave = tid >> 5;
  const int iw = i0 + wave * 16;  // this wave's 16 query rows

  __shared__ alignas(16) unsigned short ks[TILE_US];      // K tile (TDM-filled)
  __shared__ alignas(16) unsigned short vs[TILE_US];      // V^T tile (async-filled)
  __shared__ alignas(16) unsigned short pb[8 * 16 * 32];  // per-wave P scratch
  unsigned short* pw = pb + wave * 512;
  const v4u* ks4 = (const v4u*)ks;
  const v4u* vs4 = (const v4u*)vs;
  const v4u* pw4 = (const v4u*)pw;

  const unsigned ldsK = (unsigned)(uintptr_t)(&ks[0]);
  const unsigned ldsV = (unsigned)(uintptr_t)(&vs[0]);

  // Preload all Q A-fragments (K = 256 -> 8 fragments): two uint4 runs each.
  v8u qf[8];
  {
    const v4u* qrow4 =
        (const v4u*)(Qt + ((size_t)n * HW_DIM + iw + lrow) * C_DIM);
#pragma unroll
    for (int kt = 0; kt < 8; ++kt)
      qf[kt] = concat4(qrow4[kt * 4 + 2 * half], qrow4[kt * 4 + 2 * half + 1]);
  }

  v8f oa[16];
  const v8f vzero = {0.f, 0.f, 0.f, 0.f, 0.f, 0.f, 0.f, 0.f};
#pragma unroll
  for (int t = 0; t < 16; ++t) oa[t] = vzero;
  float m8[8], l8[8];
#pragma unroll
  for (int v = 0; v < 8; ++v) { m8[v] = -3.0e38f; l8[v] = 0.f; }

  const unsigned short* Kn = Kbuf + (size_t)n * 128 * TILE_US;
  const unsigned short* Vn = Vbuf + (size_t)n * 128 * TILE_US;

  for (int jb = 0; jb < HW_DIM; jb += 32) {
    const int tile = jb >> 5;
    __syncthreads();  // previous tile fully consumed
    // K tile: one TDM descriptor load issued by wave 0 (16KB contiguous image).
    if (wave == 0) {
      tdm_load_row(ldsK,
                   (unsigned long long)(uintptr_t)(Kn + (size_t)tile * TILE_US),
                   TILE_US);
    }
    // V tile: async bulk copy (ASYNCcnt path), 16B per lane per issue.
    {
      unsigned long long gV =
          (unsigned long long)(uintptr_t)(Vn + (size_t)tile * TILE_US);
      unsigned toff = (unsigned)tid * 16u;
#pragma unroll
      for (int ch = 0; ch < 4; ++ch)
        async_copy_b128(ldsV + toff + ch * 4096u, gV + toff + ch * 4096u);
    }
    if (jb + 32 < HW_DIM) {  // pull next tiles toward L2/WGP$
      __builtin_prefetch(Kn + (size_t)(tile + 1) * TILE_US + ((size_t)tid << 5), 0, 0);
      __builtin_prefetch(Vn + (size_t)(tile + 1) * TILE_US + ((size_t)tid << 5), 0, 0);
    }
    wait_async();
    if (wave == 0) __builtin_amdgcn_s_wait_tensorcnt(0);
    __syncthreads();

    // S = Q^T K  (16 x 32 scores per wave, K-dim = 256)
    v8f s[2];
#pragma unroll
    for (int nt = 0; nt < 2; ++nt) {
      v8f acc = vzero;
#pragma unroll
      for (int kt = 0; kt < 8; ++kt) {
        int fb = ((kt * 2 + half) * 32 + nt * 16 + lrow) * 2;
        acc = wmma_bf16(qf[kt], concat4(ks4[fb], ks4[fb + 1]), acc);
      }
      s[nt] = acc;
    }

    // Online softmax: each (v, lane-half) is one query row; a row's 16 columns
    // live in one 16-lane half -> width-16 xor-shuffle reductions.
    float alpha8[8];
#pragma unroll
    for (int v = 0; v < 8; ++v) {
      float mx = fmaxf(s[0][v], s[1][v]);
#pragma unroll
      for (int off = 1; off < 16; off <<= 1) mx = fmaxf(mx, __shfl_xor(mx, off, 16));
      float mnew = fmaxf(m8[v], mx);
      float a = __expf(m8[v] - mnew);
      float p0 = __expf(s[0][v] - mnew);
      float p1 = __expf(s[1][v] - mnew);
      s[0][v] = p0; s[1][v] = p1;
      float rs = p0 + p1;
#pragma unroll
      for (int off = 1; off < 16; off <<= 1) rs += __shfl_xor(rs, off, 16);
      l8[v] = l8[v] * a + rs;
      m8[v] = mnew;
      alpha8[v] = a;
    }
#pragma unroll
    for (int t = 0; t < 16; ++t)
#pragma unroll
      for (int v = 0; v < 8; ++v) oa[t][v] *= alpha8[v];

    // C/D layout -> A layout via per-wave LDS (DS ops are in-order per wave)
#pragma unroll
    for (int nt = 0; nt < 2; ++nt)
#pragma unroll
      for (int v = 0; v < 8; ++v)
        pw[(v + 8 * half) * 32 + nt * 16 + lrow] = f2bf(s[nt][v]);

    v8u a2 = concat4(pw4[lrow * 4 + 2 * half], pw4[lrow * 4 + 2 * half + 1]);

    // O += P * V^T  (K-dim = 32: one WMMA per 16-wide channel tile)
#pragma unroll
    for (int ct = 0; ct < 16; ++ct) {
      int fb = ((ct * 16 + lrow) * 2 + half) * 2;
      oa[ct] = wmma_bf16(a2, concat4(vs4[fb], vs4[fb + 1]), oa[ct]);
    }
  }

  // Epilogue: out = x + gamma * O / l
  const float g = gamma[0];
  float inv8[8];
#pragma unroll
  for (int v = 0; v < 8; ++v) inv8[v] = g / l8[v];
  const float* xn = x + (size_t)n * C_DIM * HW_DIM;
  float* on = out + (size_t)n * C_DIM * HW_DIM;
#pragma unroll
  for (int ct = 0; ct < 16; ++ct)
#pragma unroll
    for (int v = 0; v < 8; ++v) {
      int c = ct * 16 + lrow;
      int i = iw + v + 8 * half;
      size_t idx = (size_t)c * HW_DIM + i;
      on[idx] = xn[idx] + oa[ct][v] * inv8[v];
    }
}

extern "C" void kernel_launch(void* const* d_in, const int* in_sizes, int n_in,
                              void* d_out, int out_size, void* d_ws, size_t ws_size,
                              hipStream_t stream) {
  (void)in_sizes; (void)n_in; (void)out_size; (void)ws_size;
  const float* x  = (const float*)d_in[0];
  const float* Wq = (const float*)d_in[1];
  const float* bq = (const float*)d_in[2];
  const float* Wk = (const float*)d_in[3];
  const float* bk = (const float*)d_in[4];
  const float* Wv = (const float*)d_in[5];
  const float* bv = (const float*)d_in[6];
  const float* gm = (const float*)d_in[7];

  const size_t elems = (size_t)4 * HW_DIM * C_DIM;  // per tensor, bf16
  unsigned short* Qt   = (unsigned short*)d_ws;
  unsigned short* Kbuf = Qt + elems;
  unsigned short* Vbuf = Kbuf + elems;
  unsigned short* Wbf  = Vbuf + elems;  // 3 * 256 * 256 bf16

  prep_w_kernel<<<dim3(768), 256, 0, stream>>>(Wq, Wk, Wv, Wbf);
  proj_qkv_kernel<<<dim3(HW_DIM / 64, 4), 256, 0, stream>>>(
      x, Wbf, bq, bk, bv, Qt, Kbuf, Vbuf);
  flash_att_kernel<<<dim3(HW_DIM / 128, 4), 256, 0, stream>>>(
      x, gm, Qt, Kbuf, Vbuf, (float*)d_out);
}
